// SetAttentionLayer_45148696215780
// MI455X (gfx1250) — compile-verified, matrix operand order checked by reading, not run on hardware
//
#include <hip/hip_runtime.h>
#include <hip/hip_bf16.h>

typedef __attribute__((ext_vector_type(16))) _Float16 v16h;
typedef __attribute__((ext_vector_type(8)))  _Float16 v8h;
typedef __attribute__((ext_vector_type(8)))  float    v8f;

#define NROWS   500000
#define DIN     40
#define NSEG    2048
#define WIDTH   128
#define LATENT  128
#define DOTD    64
#define HEADS   4
#define KC      (DIN + LATENT)   // 168
#define WAVES   8

// ---------------- workspace layout (bytes, 64B aligned) ----------------
#define OFF_WF1     0                         // 64x128 f16 frags   : 16384
#define OFF_WF2     16384                     // 128x128 f16 frags  : 32768
#define OFF_WF3     49152                     // 128x128 f16 frags  : 32768
#define OFF_WFR     81920                     // 128x128 f16 frags  : 32768
#define OFF_WEFF    114688                    // 4*168 f32          : 2688 (pad 4096)
#define OFF_SUMS    118784                    // 2048*128 f32       : 1048576
#define OFF_COUNTS  1167360                   // 2048 f32           : 8192 (pad)
#define OFF_AGG     1175552                   // 2048*128 f32       : 1048576
#define OFF_SEGMAX  2224128                   // 2048*4 u32         : 32768 (pad)
#define OFF_SEGSUM  2256896                   // 2048*4 f32         : 32768 (pad)
#define OFF_P       2289664                   // N*4 f32            : 8000000

// k-within-tile position for the 16-bit WMMA A/B fragment layout
// (CDNA5 ISA §7.12.2). Within a lane, halves 0..7 are K = grp*8 + h and
// halves 8..15 are K = 16 + grp*8 + (h-8): two contiguous 16B runs.
__device__ __forceinline__ int klocal(int h, int lane) {
    int v = h >> 1, pos = h & 1, grp = (lane >> 4) & 1;
    return ((v & 4) ? 16 : 0) + ((v & 3) * 2 + pos) + grp * 8;
}

// Build one 16x32 f16 A fragment from a staged LDS row with two ds_load_b128.
__device__ __forceinline__ v16h load_afrag(const _Float16* rowp, int kt, int grp) {
    v8h lo = *(const v8h*)(rowp + kt * 32 + grp * 8);
    v8h hi = *(const v8h*)(rowp + kt * 32 + 16 + grp * 8);
    return __builtin_shufflevector(lo, hi, 0, 1, 2, 3, 4, 5, 6, 7,
                                   8, 9, 10, 11, 12, 13, 14, 15);
}

__device__ __forceinline__ unsigned enc_f32(float x) {
    unsigned u = __float_as_uint(x);
    return (u & 0x80000000u) ? ~u : (u | 0x80000000u);
}
__device__ __forceinline__ float dec_f32(unsigned e) {
    return __uint_as_float((e & 0x80000000u) ? (e ^ 0x80000000u) : ~e);
}

// ---------------- kernel 1: zero the accumulators ----------------
__global__ void init_kernel(float* sums, float* counts, unsigned* segmax,
                            float* segsum, int s) {
    int i = blockIdx.x * blockDim.x + threadIdx.x;
    if (i < s * LATENT) sums[i] = 0.f;
    if (i < s)          counts[i] = 0.f;
    if (i < s * HEADS) { segmax[i] = 0u; segsum[i] = 0.f; }
}

// ---------------- kernel 2: pack weights into WMMA fragments ----------------
__device__ void pack_weight(const float* __restrict__ src, _Float16* __restrict__ dst,
                            int Ksrc, int Kpad, int Nn, int tid, int nth) {
    int numKt = Kpad >> 5;
    int total = numKt * (Nn >> 4) * 512;         // halves
    for (int i = tid; i < total; i += nth) {
        int fi   = i >> 9;
        int rem  = i & 511;
        int lane = rem >> 4;
        int h    = rem & 15;
        int nt   = fi / numKt, kt = fi % numKt;
        int k    = kt * 32 + klocal(h, lane);
        int col  = nt * 16 + (lane & 15);
        dst[i] = (k < Ksrc) ? (_Float16)src[k * Nn + col] : (_Float16)0.f;
    }
}

__global__ void prep_kernel(const float* __restrict__ W1, const float* __restrict__ W2,
                            const float* __restrict__ W3, const float* __restrict__ Wr,
                            const float* __restrict__ Wk, const float* __restrict__ Wq,
                            _Float16* wf1, _Float16* wf2, _Float16* wf3, _Float16* wfr,
                            float* weff) {
    int tid = blockIdx.x * blockDim.x + threadIdx.x;
    int nth = gridDim.x * blockDim.x;
    pack_weight(W1, wf1, DIN,   64,  WIDTH,  tid, nth);
    pack_weight(W2, wf2, WIDTH, 128, WIDTH,  tid, nth);
    pack_weight(W3, wf3, WIDTH, 128, LATENT, tid, nth);
    pack_weight(Wr, wfr, LATENT,128, LATENT, tid, nth);
    // fold W_k (168x256) with W_q (4x64) and 1/sqrt(64): weff[h*KC + d]
    for (int i = tid; i < HEADS * KC; i += nth) {
        int h = i / KC, d = i % KC;
        float acc = 0.f;
        for (int j = 0; j < DOTD; ++j)
            acc += Wk[d * (HEADS * DOTD) + h * DOTD + j] * Wq[h * DOTD + j];
        weff[i] = acc * 0.125f;
    }
}

// ---------------- kernel 3: 3-layer WMMA MLP + segment-sum atomics ----------------
__global__ __launch_bounds__(256) void
mlp_kernel(const float* __restrict__ inputs, const int* __restrict__ seg,
           const float* __restrict__ b1, const float* __restrict__ b2,
           const float* __restrict__ b3,
           const _Float16* __restrict__ wf1, const _Float16* __restrict__ wf2,
           const _Float16* __restrict__ wf3,
           float* sums, float* counts, int n) {
    __shared__ _Float16 sIn [WAVES][16][64];     // 16 KB, K-padded f16 inputs
    __shared__ _Float16 sAct[WAVES][16][WIDTH];  // 32 KB
    const int tid  = threadIdx.x;
    const int w    = tid >> 5;
    const int lane = tid & 31;
    const int r0   = blockIdx.x * (WAVES * 16) + w * 16;
    const int m    = lane & 15;
    const int ncol = lane & 15;
    const int grp  = (lane >> 4) & 1;
    const int mh   = grp << 3;                   // +8 row offset for lanes 16..31

    // stage 16 input rows, converted to f16 and zero-padded K: 40 -> 64
    for (int i = lane; i < 16 * 64; i += 32) {
        int rr = i >> 6, c = i & 63;
        int row = r0 + rr;
        float v = (c < DIN && row < n) ? inputs[row * DIN + c] : 0.f;
        sIn[w][rr][c] = (_Float16)v;
    }
    // warm WGP$/L0 with the layer-2/3 weight fragment pools (global_prefetch_b8)
    __builtin_prefetch((const char*)wf2 + tid * 128, 0, 1);
    __builtin_prefetch((const char*)wf3 + tid * 128, 0, 1);
    __syncthreads();

    // ---- layer 1: [16x64(pad)] x [64x128] ----
    v16h a1[2];
#pragma unroll
    for (int kt = 0; kt < 2; ++kt)
        a1[kt] = load_afrag(&sIn[w][m][0], kt, grp);
#pragma unroll
    for (int nt = 0; nt < 8; ++nt) {
        v8f acc = {};
#pragma unroll
        for (int kt = 0; kt < 2; ++kt) {
            v16h b = *(const v16h*)(wf1 + ((nt * 2 + kt) * 512) + lane * 16);
            acc = __builtin_amdgcn_wmma_f32_16x16x32_f16(false, a1[kt], false, b,
                                                         (short)0, acc, false, false);
        }
        float bias = b1[nt * 16 + ncol];
#pragma unroll
        for (int r = 0; r < 8; ++r)
            sAct[w][r + mh][nt * 16 + ncol] = (_Float16)fmaxf(acc[r] + bias, 0.f);
    }
    __syncthreads();

    // ---- layer 2: [16x128] x [128x128] ----
    v16h a2[4];
#pragma unroll
    for (int kt = 0; kt < 4; ++kt)
        a2[kt] = load_afrag(&sAct[w][m][0], kt, grp);
    __syncthreads();
#pragma unroll
    for (int nt = 0; nt < 8; ++nt) {
        v8f acc = {};
#pragma unroll
        for (int kt = 0; kt < 4; ++kt) {
            v16h b = *(const v16h*)(wf2 + ((nt * 4 + kt) * 512) + lane * 16);
            acc = __builtin_amdgcn_wmma_f32_16x16x32_f16(false, a2[kt], false, b,
                                                         (short)0, acc, false, false);
        }
        float bias = b2[nt * 16 + ncol];
#pragma unroll
        for (int r = 0; r < 8; ++r)
            sAct[w][r + mh][nt * 16 + ncol] = (_Float16)fmaxf(acc[r] + bias, 0.f);
    }
    __syncthreads();

    // ---- layer 3: [16x128] x [128x128], atomically scatter into segment sums ----
    v16h a3[4];
#pragma unroll
    for (int kt = 0; kt < 4; ++kt)
        a3[kt] = load_afrag(&sAct[w][m][0], kt, grp);
#pragma unroll
    for (int nt = 0; nt < 8; ++nt) {
        v8f acc = {};
#pragma unroll
        for (int kt = 0; kt < 4; ++kt) {
            v16h b = *(const v16h*)(wf3 + ((nt * 4 + kt) * 512) + lane * 16);
            acc = __builtin_amdgcn_wmma_f32_16x16x32_f16(false, a3[kt], false, b,
                                                         (short)0, acc, false, false);
        }
        float bias = b3[nt * 16 + ncol];
#pragma unroll
        for (int r = 0; r < 8; ++r) {
            int row = r0 + r + mh;
            if (row < n) {
                float v = fmaxf(acc[r] + bias, 0.f);
                atomicAdd(&sums[seg[row] * LATENT + nt * 16 + ncol], v);
            }
        }
    }
    if (lane < 16) {
        int row = r0 + lane;
        if (row < n) atomicAdd(&counts[seg[row]], 1.0f);
    }
}

// ---------------- kernel 4: rho — mean, 128x128 WMMA GEMM, ReLU ----------------
__global__ __launch_bounds__(256) void
rho_kernel(const float* __restrict__ sums, const float* __restrict__ counts,
           const float* __restrict__ br, const _Float16* __restrict__ wfr,
           float* agg, int s) {
    __shared__ _Float16 sA[WAVES][16][LATENT];   // 32 KB
    const int tid  = threadIdx.x;
    const int w    = tid >> 5;
    const int lane = tid & 31;
    const int r0   = blockIdx.x * (WAVES * 16) + w * 16;
    const int m    = lane & 15;
    const int ncol = lane & 15;
    const int grp  = (lane >> 4) & 1;
    const int mh   = grp << 3;

    for (int i = lane; i < 16 * LATENT; i += 32) {
        int rr = i >> 7, c = i & 127;
        int row = r0 + rr;
        float v = 0.f;
        if (row < s) v = sums[row * LATENT + c] / fmaxf(counts[row], 1.f);
        sA[w][rr][c] = (_Float16)v;
    }
    __syncthreads();

    v16h a[4];
#pragma unroll
    for (int kt = 0; kt < 4; ++kt)
        a[kt] = load_afrag(&sA[w][m][0], kt, grp);
#pragma unroll
    for (int nt = 0; nt < 8; ++nt) {
        v8f acc = {};
#pragma unroll
        for (int kt = 0; kt < 4; ++kt) {
            v16h b = *(const v16h*)(wfr + ((nt * 4 + kt) * 512) + lane * 16);
            acc = __builtin_amdgcn_wmma_f32_16x16x32_f16(false, a[kt], false, b,
                                                         (short)0, acc, false, false);
        }
        float bias = br[nt * 16 + ncol];
#pragma unroll
        for (int r = 0; r < 8; ++r) {
            int row = r0 + r + mh;
            if (row < s)
                agg[row * LATENT + nt * 16 + ncol] = fmaxf(acc[r] + bias, 0.f);
        }
    }
}

// ---------------- kernel 5: preattn GEMV (W_k folded with W_q) + segment max ----
__global__ __launch_bounds__(256) void
preattn_kernel(const float* __restrict__ inputs, const int* __restrict__ seg,
               const float* __restrict__ agg, const float* __restrict__ weff,
               float* p, unsigned* segmax, int n) {
    __shared__ float sw[HEADS][KC];
    for (int i = threadIdx.x; i < HEADS * KC; i += blockDim.x)
        sw[i / KC][i % KC] = weff[i];
    __syncthreads();
    int row = blockIdx.x * blockDim.x + threadIdx.x;
    if (row >= n) return;
    int s = seg[row];
    float acc[HEADS] = {0.f, 0.f, 0.f, 0.f};
    const float* in = inputs + (long)row * DIN;
    for (int d = 0; d < DIN; ++d) {
        float x = in[d];
#pragma unroll
        for (int h = 0; h < HEADS; ++h) acc[h] += x * sw[h][d];
    }
    const float* ag = agg + (long)s * LATENT;
    for (int l = 0; l < LATENT; ++l) {
        float x = ag[l];
#pragma unroll
        for (int h = 0; h < HEADS; ++h) acc[h] += x * sw[h][DIN + l];
    }
#pragma unroll
    for (int h = 0; h < HEADS; ++h) {
        p[row * HEADS + h] = acc[h];
        atomicMax(&segmax[s * HEADS + h], enc_f32(acc[h]));
    }
}

// ---------------- kernel 6: exp(p - max) + segment denom ----------------
__global__ void expsum_kernel(const int* __restrict__ seg, const float* __restrict__ p,
                              const unsigned* __restrict__ segmax, float* segsum,
                              float* out, int n) {
    int row = blockIdx.x * blockDim.x + threadIdx.x;
    if (row >= n) return;
    int s = seg[row];
#pragma unroll
    for (int h = 0; h < HEADS; ++h) {
        float mx = dec_f32(segmax[s * HEADS + h]);
        float e  = expf(p[row * HEADS + h] - mx);
        out[(long)h * n + row] = e;
        atomicAdd(&segsum[s * HEADS + h], e);
    }
}

// ---------------- kernel 7: normalize ----------------
__global__ void norm_kernel(const int* __restrict__ seg, const float* __restrict__ segsum,
                            float* out, int n) {
    int row = blockIdx.x * blockDim.x + threadIdx.x;
    if (row >= n) return;
    int s = seg[row];
#pragma unroll
    for (int h = 0; h < HEADS; ++h)
        out[(long)h * n + row] /= segsum[s * HEADS + h];
}

extern "C" void kernel_launch(void* const* d_in, const int* in_sizes, int n_in,
                              void* d_out, int out_size, void* d_ws, size_t ws_size,
                              hipStream_t stream) {
    const float* inputs = (const float*)d_in[0];
    const int*   seg    = (const int*)  d_in[1];
    // d_in[2] = lengths (unused by the reference math)
    const float* W1 = (const float*)d_in[3];
    const float* b1 = (const float*)d_in[4];
    const float* W2 = (const float*)d_in[5];
    const float* b2 = (const float*)d_in[6];
    const float* W3 = (const float*)d_in[7];
    const float* b3 = (const float*)d_in[8];
    const float* Wr = (const float*)d_in[9];
    const float* br = (const float*)d_in[10];
    const float* Wk = (const float*)d_in[11];
    const float* Wq = (const float*)d_in[12];
    float* out = (float*)d_out;

    const int n = in_sizes[0] / DIN;     // 500000
    const int s = in_sizes[2];           // 2048

    char* ws = (char*)d_ws;
    _Float16* wf1   = (_Float16*)(ws + OFF_WF1);
    _Float16* wf2   = (_Float16*)(ws + OFF_WF2);
    _Float16* wf3   = (_Float16*)(ws + OFF_WF3);
    _Float16* wfr   = (_Float16*)(ws + OFF_WFR);
    float*    weff  = (float*)   (ws + OFF_WEFF);
    float*    sums  = (float*)   (ws + OFF_SUMS);
    float*    cnts  = (float*)   (ws + OFF_COUNTS);
    float*    agg   = (float*)   (ws + OFF_AGG);
    unsigned* smax  = (unsigned*)(ws + OFF_SEGMAX);
    float*    ssum  = (float*)   (ws + OFF_SEGSUM);
    float*    p     = (float*)   (ws + OFF_P);

    // 1. zero accumulators
    {
        int tot = s * LATENT;
        init_kernel<<<(tot + 255) / 256, 256, 0, stream>>>(sums, cnts, smax, ssum, s);
    }
    // 2. pack weights + fold W_k*W_q
    prep_kernel<<<64, 256, 0, stream>>>(W1, W2, W3, Wr, Wk, Wq,
                                        wf1, wf2, wf3, wfr, weff);
    // 3. WMMA MLP + segment sums
    mlp_kernel<<<(n + WAVES * 16 - 1) / (WAVES * 16), 256, 0, stream>>>(
        inputs, seg, b1, b2, b3, wf1, wf2, wf3, sums, cnts, n);
    // 4. rho GEMM
    rho_kernel<<<(s + WAVES * 16 - 1) / (WAVES * 16), 256, 0, stream>>>(
        sums, cnts, br, wfr, agg, s);
    // 5. preattn + segment max
    preattn_kernel<<<(n + 255) / 256, 256, 0, stream>>>(inputs, seg, agg, weff,
                                                        p, smax, n);
    // 6. exp + segment denom
    expsum_kernel<<<(n + 255) / 256, 256, 0, stream>>>(seg, p, smax, ssum, out, n);
    // 7. normalize
    norm_kernel<<<(n + 255) / 256, 256, 0, stream>>>(seg, ssum, out, n);
}
